// TransformerComputeLayer_32169305047324
// MI455X (gfx1250) — matrix operand order repaired and checked by reading, not compiled, not run on hardware
//
#include <hip/hip_runtime.h>

// ---------------------------------------------------------------------------
// Transformer layer for MI455X (gfx1250, wave32, WMMA).
// fp32 I/O, bf16 WMMA (v_wmma_f32_16x16x32_bf16) with fp32 accumulation.
// GEMMs: double-buffered LDS staging via async global->LDS (ASYNCcnt) when
// the toolchain exposes the gfx1250 builtin; sync b128 copies otherwise.
// ---------------------------------------------------------------------------

typedef __bf16 bf16;
typedef bf16  v8bf  __attribute__((ext_vector_type(8)));
typedef bf16  v16bf __attribute__((ext_vector_type(16)));
typedef float v8f   __attribute__((ext_vector_type(8)));
typedef int   v4i_  __attribute__((vector_size(16)));   // type the async builtin wants

#define DEV __device__ __forceinline__

#if __has_builtin(__builtin_amdgcn_global_load_async_to_lds_b128)
#define HAS_ASYNC_LDS 1
#else
#define HAS_ASYNC_LDS 0
#endif

// 16-byte global -> LDS copy (async DMA path if available)
DEV void cp16(bf16* l, const bf16* g) {
#if HAS_ASYNC_LDS
    __builtin_amdgcn_global_load_async_to_lds_b128((v4i_*)g, (v4i_*)l, 0, 0);
#else
    *(uint4*)l = *(const uint4*)g;
#endif
}
DEV void cp_wait() {
#if HAS_ASYNC_LDS
    asm volatile("s_wait_asynccnt 0" ::: "memory");
#endif
}

// Build a 16-element bf16 fragment from two 16-byte-aligned 8-element chunks.
DEV v16bf ldfrag(const bf16* p0, const bf16* p1) {
    v8bf a = *(const v8bf*)p0;
    v8bf b = *(const v8bf*)p1;
    v16bf r;
#pragma unroll
    for (int i = 0; i < 8; ++i) { r[i] = a[i]; r[i + 8] = b[i]; }
    return r;
}

DEV v8f wmma_bf16(v16bf a, v16bf b, v8f c) {
    // (neg_a, A, neg_b, B, c_mod, C, reuse_a, reuse_b)
    return __builtin_amdgcn_wmma_f32_16x16x32_bf16(false, a, false, b,
                                                   (short)0, c, false, false);
}

// ---------------------------------------------------------------------------
// Constants for this problem instance
// ---------------------------------------------------------------------------
#define B_    2
#define S_    2048
#define D_    1024
#define H_    16
#define HD_   64
#define FF_   4096
#define ROWS_ (B_ * S_)          // 4096
#define NQKV_ (3 * D_)           // 3072

// ---------------------------------------------------------------------------
// Weight conversion kernels (fp32 -> bf16, transposed to N x K "BT" layout)
// ---------------------------------------------------------------------------
__global__ void cvt_qkv_k(const float* __restrict__ W, bf16* __restrict__ BT) {
    int i = blockIdx.x * 256 + threadIdx.x;     // i over N*K = 3072*1024
    int n = i >> 10, d = i & 1023;              // n = g*64 + h-col
    int g = n >> 6, hc = n & 63;
    BT[i] = (bf16)W[((size_t)g * D_ + d) * HD_ + hc];
}

__global__ void cvt_T_k(const float* __restrict__ W, bf16* __restrict__ BT,
                        int Kd, int Nd) {
    int i = blockIdx.x * 256 + threadIdx.x;     // i over N*K
    int n = i / Kd, k = i % Kd;
    BT[i] = (bf16)W[(size_t)k * Nd + n];
}

// ---------------------------------------------------------------------------
// LayerNorm (D=1024) + cast to bf16.  One block (8 waves) per row.
// ---------------------------------------------------------------------------
__global__ __launch_bounds__(256) void ln_cast_k(const float* __restrict__ x,
                                                 const float* __restrict__ g,
                                                 const float* __restrict__ b,
                                                 bf16* __restrict__ out) {
    const int row = blockIdx.x;
    const int t = threadIdx.x, wid = t >> 5, lane = t & 31;
    const float* xr = x + (size_t)row * D_;
    float v[4], s = 0.f, sq = 0.f;
#pragma unroll
    for (int i = 0; i < 4; ++i) {
        v[i] = xr[t + i * 256];
        s += v[i]; sq += v[i] * v[i];
    }
#pragma unroll
    for (int o = 16; o > 0; o >>= 1) {
        s  += __shfl_xor(s,  o, 32);
        sq += __shfl_xor(sq, o, 32);
    }
    __shared__ float rs[8], rq[8];
    if (lane == 0) { rs[wid] = s; rq[wid] = sq; }
    __syncthreads();
    s = 0.f; sq = 0.f;
#pragma unroll
    for (int w = 0; w < 8; ++w) { s += rs[w]; sq += rq[w]; }
    const float mu  = s * (1.f / D_);
    const float var = sq * (1.f / D_) - mu * mu;
    const float inv = rsqrtf(var + 1e-5f);
#pragma unroll
    for (int i = 0; i < 4; ++i) {
        int c = t + i * 256;
        out[(size_t)row * D_ + c] = (bf16)((v[i] - mu) * inv * g[c] + b[c]);
    }
}

// ---------------------------------------------------------------------------
// Tiled bf16 GEMM:  C[M,N] = A[M,K] * BT[N,K]^T  (+ bias, epilogue by MODE)
//   MODE 0: out bf16, + side-write of V rows transposed into vT
//   MODE 1: out fp32 = resid + C
//   MODE 2: out bf16 = relu(C)
// Block tile 128 x (NWT*64); Kc=32, double-buffered async LDS staging.
// 8 waves in a 2x4 grid; wave tile 64 x (NWT*16) -> 4*NWT wmma per K-step.
// ---------------------------------------------------------------------------
template <int MODE, int NWT>
__global__ __launch_bounds__(256) void gemm_bf16_k(
    const bf16* __restrict__ A, const bf16* __restrict__ BT,
    const float* __restrict__ bias,
    bf16* __restrict__ outB, float* __restrict__ outF,
    const float* __restrict__ resid, bf16* __restrict__ vT,
    int M, int N, int K) {
    constexpr int BN = NWT * 64;                 // block N extent
    __shared__ __align__(16) bf16 lA[2][128][40];   // +8 pad: 80B row stride
    __shared__ __align__(16) bf16 lB[2][BN][40];

    const int t = threadIdx.x;
    const int wid = t >> 5, lane = t & 31;
    const int hb = lane >> 4, lr = lane & 15;
    const int wm = wid >> 2, wn = wid & 3;       // 2 x 4 wave grid
    const int blockM = blockIdx.y * 128, blockN = blockIdx.x * BN;

    v8f acc[4][NWT] = {};

    auto load_tile = [&](int buf, int k0) {
#pragma unroll
        for (int i = 0; i < 2; ++i) {            // A: 128 rows * 4 chunks
            int c = t + i * 256;
            int row = c >> 2, kc = (c & 3) * 8;
            cp16(&lA[buf][row][kc], A + (size_t)(blockM + row) * K + k0 + kc);
        }
#pragma unroll
        for (int i = 0; i < NWT; ++i) {          // B: BN rows * 4 chunks
            int c = t + i * 256;
            int row = c >> 2, kc = (c & 3) * 8;
            cp16(&lB[buf][row][kc], BT + (size_t)(blockN + row) * K + k0 + kc);
        }
    };

    load_tile(0, 0);
    cp_wait();
    __syncthreads();

    int buf = 0;
    for (int k0 = 0; k0 < K; k0 += 32) {
        if (k0 + 32 < K) load_tile(buf ^ 1, k0 + 32);   // overlap DMA w/ WMMA

        v16bf bfr[NWT];
#pragma unroll
        for (int nt = 0; nt < NWT; ++nt) {
            const bf16* br = &lB[buf][wn * (NWT * 16) + nt * 16 + lr][hb * 16];
            bfr[nt] = ldfrag(br, br + 8);
        }
#pragma unroll
        for (int mt = 0; mt < 4; ++mt) {
            const bf16* ar = &lA[buf][wm * 64 + mt * 16 + lr][0];
            v16bf afr = ldfrag(ar + hb * 16, ar + hb * 16 + 8);
#pragma unroll
            for (int nt = 0; nt < NWT; ++nt)
                acc[mt][nt] = wmma_bf16(afr, bfr[nt], acc[mt][nt]);
        }
        cp_wait();
        __syncthreads();
        buf ^= 1;
    }

    // Epilogue: C layout => element (vgpr r, lane): M = r + 8*hb, N = lr
#pragma unroll
    for (int mt = 0; mt < 4; ++mt)
#pragma unroll
        for (int nt = 0; nt < NWT; ++nt)
#pragma unroll
            for (int r = 0; r < 8; ++r) {
                int gm = blockM + wm * 64 + mt * 16 + r + hb * 8;
                int gn = blockN + wn * (NWT * 16) + nt * 16 + lr;
                float val = acc[mt][nt][r] + bias[gn];
                if constexpr (MODE == 0) {
                    outB[(size_t)gm * N + gn] = (bf16)val;
                    if (gn >= 2 * D_) {          // V part -> transposed copy
                        int bb = gm >> 11, s = gm & (S_ - 1);
                        int gg = gn - 2 * D_;
                        int hh = gg >> 6, dd = gg & 63;
                        vT[(((size_t)(bb * H_ + hh) * HD_ + dd) << 11) + s] = (bf16)val;
                    }
                } else if constexpr (MODE == 1) {
                    outF[(size_t)gm * N + gn] = resid[(size_t)gm * N + gn] + val;
                } else {
                    outB[(size_t)gm * N + gn] = (bf16)fmaxf(val, 0.f);
                }
            }
}

// ---------------------------------------------------------------------------
// Flash-style attention. Grid (S/128, H, B); 8 waves, 16 q-rows per wave.
// scores = Q*K^T/8 (WMMA), online softmax. Row-max via xor-shuffle tree on the
// half-wave; row-SUM via one extra WMMA against an all-ones B fragment
// (matrix pipe instead of 32 ds_bpermute's). P*V via WMMA from the vT copy.
// ---------------------------------------------------------------------------
__global__ __launch_bounds__(256) void attn_k(const bf16* __restrict__ qkv,
                                              const bf16* __restrict__ vT,
                                              bf16* __restrict__ out) {
    const int qt = blockIdx.x, h = blockIdx.y, b = blockIdx.z;
    const int t = threadIdx.x, wid = t >> 5, lane = t & 31;
    const int hb = lane >> 4, lr = lane & 15;
    const size_t rbase = (size_t)b * S_;         // row base in (B*S, 3072)
    const int q0 = qt * 128 + wid * 16;

    __shared__ __align__(16) bf16 pL[8][16][40];

    // Q fragments (A layout): rows s, K = d (64) -> two K=32 fragments
    v16bf qf[2];
#pragma unroll
    for (int kd = 0; kd < 2; ++kd) {
        const bf16* qr = qkv + (rbase + q0 + lr) * NQKV_ + h * HD_ + kd * 32;
        qf[kd] = ldfrag(qr + hb * 16, qr + hb * 16 + 8);
    }

    v16bf ones;
#pragma unroll
    for (int i = 0; i < 16; ++i) ones[i] = (bf16)1.0f;

    float Mx[8], Ls[8];
    v8f O[4] = {};
#pragma unroll
    for (int r = 0; r < 8; ++r) { Mx[r] = -1e30f; Ls[r] = 0.f; }

    for (int j0 = 0; j0 < S_; j0 += 32) {
        // ---- scores: two 16x16 tiles over K=64
        v8f sc[2] = {};
#pragma unroll
        for (int jt = 0; jt < 2; ++jt)
#pragma unroll
            for (int kd = 0; kd < 2; ++kd) {
                const bf16* kr = qkv + (rbase + j0 + jt * 16 + lr) * NQKV_ +
                                 D_ + h * HD_ + kd * 32;
                v16bf kf = ldfrag(kr + hb * 16, kr + hb * 16 + 8);
                sc[jt] = wmma_bf16(qf[kd], kf, sc[jt]);
            }
        // ---- online max (rows live on one half-wave: reduce masks 1..8)
        float scl[8];
#pragma unroll
        for (int r = 0; r < 8; ++r) {
            sc[0][r] *= 0.125f; sc[1][r] *= 0.125f;
            float tm = fmaxf(sc[0][r], sc[1][r]);
            tm = fmaxf(tm, __shfl_xor(tm, 1, 32));
            tm = fmaxf(tm, __shfl_xor(tm, 2, 32));
            tm = fmaxf(tm, __shfl_xor(tm, 4, 32));
            tm = fmaxf(tm, __shfl_xor(tm, 8, 32));
            float nM = fmaxf(Mx[r], tm);
            scl[r] = __expf(Mx[r] - nM);
            Mx[r] = nM;
        }
#pragma unroll
        for (int dt = 0; dt < 4; ++dt)
#pragma unroll
            for (int r = 0; r < 8; ++r) O[dt][r] *= scl[r];

        __syncthreads();
#pragma unroll
        for (int r = 0; r < 8; ++r) {
            float p0 = __expf(sc[0][r] - Mx[r]);
            float p1 = __expf(sc[1][r] - Mx[r]);
            pL[wid][r + hb * 8][lr]      = (bf16)p0;   // C layout -> row-major
            pL[wid][r + hb * 8][16 + lr] = (bf16)p1;
        }
        __syncthreads();

        // ---- P (A layout from LDS); row-sum on the matrix pipe (P x ones)
        const bf16* pr = &pL[wid][lr][0];
        v16bf pf = ldfrag(pr + hb * 16, pr + hb * 16 + 8);
        v8f rsum = {};
        rsum = wmma_bf16(pf, ones, rsum);
#pragma unroll
        for (int r = 0; r < 8; ++r) Ls[r] = Ls[r] * scl[r] + rsum[r];

        // ---- O += P x V (B layout from vT), K = 32
#pragma unroll
        for (int dt = 0; dt < 4; ++dt) {
            const bf16* vr = vT +
                (((size_t)(b * H_ + h) * HD_ + dt * 16 + lr) << 11) + j0 + hb * 16;
            v16bf vf = ldfrag(vr, vr + 8);
            O[dt] = wmma_bf16(pf, vf, O[dt]);
        }
    }

    // ---- normalize + store bf16 into (B*S, 1024) with head-major columns
#pragma unroll
    for (int dt = 0; dt < 4; ++dt)
#pragma unroll
        for (int r = 0; r < 8; ++r) {
            int sr = q0 + r + hb * 8;
            out[(rbase + sr) * D_ + h * HD_ + dt * 16 + lr] =
                (bf16)(O[dt][r] / Ls[r]);
        }
}

// ---------------------------------------------------------------------------
// Launch
// ---------------------------------------------------------------------------
extern "C" void kernel_launch(void* const* d_in, const int* in_sizes, int n_in,
                              void* d_out, int out_size, void* d_ws, size_t ws_size,
                              hipStream_t stream) {
    const float* x     = (const float*)d_in[0];
    const float* Wqkv  = (const float*)d_in[1];
    const float* bqkv  = (const float*)d_in[2];
    const float* Wo    = (const float*)d_in[3];
    const float* bo    = (const float*)d_in[4];
    const float* W1    = (const float*)d_in[5];
    const float* b1    = (const float*)d_in[6];
    const float* W2    = (const float*)d_in[7];
    const float* b2    = (const float*)d_in[8];
    const float* ln1g  = (const float*)d_in[9];
    const float* ln1b  = (const float*)d_in[10];
    const float* ln2g  = (const float*)d_in[11];
    const float* ln2b  = (const float*)d_in[12];
    (void)in_sizes; (void)n_in; (void)out_size; (void)ws_size;

    char* ws = (char*)d_ws;
    size_t off = 0;
    auto take = [&](size_t bytes) { char* p = ws + off; off += (bytes + 255) & ~size_t(255); return p; };

    bf16*  BTqkv = (bf16*)take((size_t)NQKV_ * D_ * 2);     //  6.0 MB
    bf16*  BTo   = (bf16*)take((size_t)D_ * D_ * 2);        //  2.0 MB
    bf16*  BT1   = (bf16*)take((size_t)FF_ * D_ * 2);       //  8.0 MB
    bf16*  BT2   = (bf16*)take((size_t)D_ * FF_ * 2);       //  8.0 MB
    bf16*  normed= (bf16*)take((size_t)ROWS_ * D_ * 2);     //  8.0 MB (reused for LN2)
    bf16*  qkvB  = (bf16*)take((size_t)ROWS_ * NQKV_ * 2);  // 24.0 MB
    bf16*  vTb   = (bf16*)take((size_t)B_ * H_ * HD_ * S_ * 2); // 8.0 MB
    bf16*  attnB = (bf16*)take((size_t)ROWS_ * D_ * 2);     //  8.0 MB
    float* x1    = (float*)take((size_t)ROWS_ * D_ * 4);    // 16.0 MB
    bf16*  hB    = (bf16*)take((size_t)ROWS_ * FF_ * 2);    // 32.0 MB

    // 1) weights -> bf16, transposed N x K
    cvt_qkv_k<<<(NQKV_ * D_) / 256, 256, 0, stream>>>(Wqkv, BTqkv);
    cvt_T_k<<<(D_ * D_) / 256, 256, 0, stream>>>(Wo, BTo, D_, D_);
    cvt_T_k<<<(D_ * FF_) / 256, 256, 0, stream>>>(W1, BT1, D_, FF_);
    cvt_T_k<<<(FF_ * D_) / 256, 256, 0, stream>>>(W2, BT2, FF_, D_);

    // 2) LN1 + cast
    ln_cast_k<<<ROWS_, 256, 0, stream>>>(x, ln1g, ln1b, normed);

    // 3) fused QKV GEMM (4096 x 3072 x 1024), wide tiles, + V-transpose write
    gemm_bf16_k<0, 4><<<dim3(NQKV_ / 256, ROWS_ / 128), 256, 0, stream>>>(
        normed, BTqkv, bqkv, qkvB, nullptr, nullptr, vTb, ROWS_, NQKV_, D_);

    // 4) attention (flash-style, WMMA for QK^T, row-sum and PV)
    attn_k<<<dim3(S_ / 128, H_, B_), 256, 0, stream>>>(qkvB, vTb, attnB);

    // 5) output projection + residual -> x1 (fp32); narrow tiles for occupancy
    gemm_bf16_k<1, 2><<<dim3(D_ / 128, ROWS_ / 128), 256, 0, stream>>>(
        attnB, BTo, bo, nullptr, x1, x, nullptr, ROWS_, D_, D_);

    // 6) LN2 + cast (reuse `normed`)
    ln_cast_k<<<ROWS_, 256, 0, stream>>>(x1, ln2g, ln2b, normed);

    // 7) FF1 GEMM + ReLU (4096 x 4096 x 1024), wide tiles
    gemm_bf16_k<2, 4><<<dim3(FF_ / 256, ROWS_ / 128), 256, 0, stream>>>(
        normed, BT1, b1, hB, nullptr, nullptr, nullptr, ROWS_, FF_, D_);

    // 8) FF2 GEMM + bias + residual -> d_out (4096 x 1024 x 4096)
    gemm_bf16_k<1, 2><<<dim3(D_ / 128, ROWS_ / 128), 256, 0, stream>>>(
        hB, BT2, b2, nullptr, (float*)d_out, x1, nullptr, ROWS_, D_, FF_);
}
